// MPNN_A_15161234555431
// MI455X (gfx1250) — matrix operand order repaired and checked by reading, not compiled, not run on hardware
//
#include <hip/hip_runtime.h>
#include <hip/hip_bf16.h>

// ---------------------------------------------------------------------------
// MI455X (gfx1250) 2-layer edge-attention MPNN.
// GEMMs: v_wmma_f32_16x16x32_bf16, wave32. Wave tile = 32 rows x 64 cols
// (8 accumulators, each B fragment reused 2x with the reuse_b hint).
// KP/LDA are template constants -> fully unrolled K loop, all loads are
// base+immediate (no loop-carried address VALU -> no WMMA hazard NOPs).
// ---------------------------------------------------------------------------

typedef __attribute__((ext_vector_type(16))) __bf16 v16bf;
typedef __attribute__((ext_vector_type(8)))  __bf16 v8bf;
typedef __attribute__((ext_vector_type(8)))  float  v8f;

#define LRELU_SLOPE 0.2f
#define BN_EPS 1e-5f

// ---------------------------------------------------------------------------
// Pack f32 weights [K,N] row-major into WMMA B-fragment layout:
//   out[(((nb*Kb)+kb)*32 + lane)*16 + j]
//   lane<16 : k = kb*32 + (j<8 ? j : 8+j),  n = nb*16 + lane
//   lane>=16: k = kb*32 + (j<8 ? 8+j : 16+j), n = nb*16 + (lane-16)
// Tile order is nb-major, so packs for different column groups concatenate.
// ---------------------------------------------------------------------------
__global__ void k_pack_w(const float* __restrict__ W, int K, int N,
                         __bf16* __restrict__ out, int Kb, int Nb)
{
  const int i = blockIdx.x * blockDim.x + threadIdx.x;
  if (i >= Kb * Nb * 512) return;
  const int j    = i & 15;
  const int lane = (i >> 4) & 31;
  const int tid  = i >> 9;
  const int kb   = tid % Kb;
  const int nb   = tid / Kb;
  const int n = nb * 16 + (lane & 15);
  const int k = kb * 32 + (j >> 3) * 16 + (lane >> 4) * 8 + (j & 7);
  out[i] = (k < K && n < N) ? (__bf16)W[(size_t)k * N + n] : (__bf16)0.0f;
}

// ---------------------------------------------------------------------------
// GEMM: C[M,N] = act(A @ W + bias[N]); A bf16 with compile-time row stride
// LDA and depth KP (KP%32==0, rows allocated to ceil32(M)); C row stride ldc.
// ---------------------------------------------------------------------------
template <int OUT_BF16, int ACT, int KP, int LDA>
__global__ __launch_bounds__(128) void gemm_bf16(
    const __bf16* __restrict__ A, const __bf16* __restrict__ Bp,
    const float* __restrict__ bias, void* __restrict__ Cout,
    int M, int N, int ldc)
{
  constexpr int Kb = KP / 32;
  const int lane = threadIdx.x & 31;
  const int wave = threadIdx.x >> 5;
  const int row0 = (blockIdx.x * 4 + wave) * 32;
  if (row0 >= M) return;
  const int col0 = blockIdx.y * 64;
  const int nb0  = blockIdx.y * 4;
  const int half = lane >> 4;

  v8f acc[8];
  const v8f vzero = {0.f, 0.f, 0.f, 0.f, 0.f, 0.f, 0.f, 0.f};
#pragma unroll
  for (int t = 0; t < 8; ++t) acc[t] = vzero;

  const __bf16* __restrict__ A0 =
      A + (size_t)(row0 + (lane & 15)) * LDA + half * 8;
  const __bf16* __restrict__ A1 = A0 + (size_t)16 * LDA;
  const __bf16* __restrict__ Bl = Bp + ((size_t)nb0 * Kb * 32 + lane) * 16;

#pragma unroll
  for (int kb = 0; kb < Kb; ++kb) {
    union { v16bf v; v8bf h[2]; } a0u, a1u;
    a0u.h[0] = *(const v8bf*)(A0 + kb * 32);        // immediate offsets only
    a0u.h[1] = *(const v8bf*)(A0 + kb * 32 + 16);
    a1u.h[0] = *(const v8bf*)(A1 + kb * 32);
    a1u.h[1] = *(const v8bf*)(A1 + kb * 32 + 16);
    v16bf bfrag[4];
#pragma unroll
    for (int t = 0; t < 4; ++t)
      bfrag[t] = *(const v16bf*)(Bl + (t * Kb + kb) * 512);
#pragma unroll
    for (int t = 0; t < 4; ++t) {
      acc[t] = __builtin_amdgcn_wmma_f32_16x16x32_bf16(
          false, a0u.v, false, bfrag[t], (short)0, acc[t], false, false);
      // same B as previous identical instruction -> reuse_b hint legal
      acc[4 + t] = __builtin_amdgcn_wmma_f32_16x16x32_bf16(
          false, a1u.v, false, bfrag[t], (short)0, acc[4 + t], false, true);
    }
  }

  // epilogue: bias + activation + store (guards only here)
#pragma unroll
  for (int g = 0; g < 2; ++g) {
#pragma unroll
    for (int t = 0; t < 4; ++t) {
#pragma unroll
      for (int r = 0; r < 8; ++r) {
        const int m = row0 + g * 16 + r + half * 8;
        const int n = col0 + t * 16 + (lane & 15);
        if (m < M && n < N) {
          float v = acc[g * 4 + t][r] + bias[n];
          if (ACT == 1) v = (v > 0.f) ? v : LRELU_SLOPE * v;
          if (OUT_BF16)
            ((__bf16*)Cout)[(size_t)m * ldc + n] = (__bf16)v;
          else
            ((float*)Cout)[(size_t)m * ldc + n] = v;
        }
      }
    }
  }
}

// ---------------------------------------------------------------------------
// Elementwise / reduction kernels
// ---------------------------------------------------------------------------

// f32 [M,K] -> bf16 [Mp,Kp], zero padding
__global__ void k_cvt_pad(const float* __restrict__ src,
                          __bf16* __restrict__ dst, int M, int K, int Mp,
                          int Kp)
{
  const int i = blockIdx.x * blockDim.x + threadIdx.x;
  if (i >= Mp * Kp) return;
  const int r = i / Kp, c = i % Kp;
  dst[i] = (r < M && c < K) ? (__bf16)src[(size_t)r * K + c] : (__bf16)0.0f;
}

// node = v_emb[node_fea]; 8 contiguous dims per thread (vector ld/st)
__global__ void k_embed_nodes(const int* __restrict__ fea,
                              const float* __restrict__ emb,
                              float* __restrict__ nodef,
                              __bf16* __restrict__ nodeb, int nn)
{
  const int i = blockIdx.x * blockDim.x + threadIdx.x;
  if (i >= nn * 8) return;
  const int v = i >> 3, c = (i & 7) * 8;
  const float* src = emb + (size_t)fea[v] * 64 + c;
  float*  nf = nodef + (size_t)v * 64 + c;
  __bf16* nb = nodeb + (size_t)v * 64 + c;
  v8bf bb;
#pragma unroll
  for (int j = 0; j < 8; ++j) {
    const float x = src[j];
    nf[j] = x;
    bb[j] = (__bf16)x;
  }
  *(v8bf*)nb = bb;
}

// z = concat(node[idx1], node[idx2], edge) -> bf16 [NE,192], 16B chunks
__global__ void k_build_z(const __bf16* __restrict__ nodeb,
                          const float* __restrict__ edgef,
                          const int* __restrict__ idx1,
                          const int* __restrict__ idx2,
                          __bf16* __restrict__ z, int ne)
{
  const int i = blockIdx.x * blockDim.x + threadIdx.x;  // ne * 24 chunks
  if (i >= ne * 24) return;
  const int e = i / 24, c = i % 24;
  v8bf val;
  if (c < 8) {
    val = *(const v8bf*)(nodeb + (size_t)idx1[e] * 64 + c * 8);
  } else if (c < 16) {
    val = *(const v8bf*)(nodeb + (size_t)idx2[e] * 64 + (c - 8) * 8);
  } else {
    const float* ep = edgef + (size_t)e * 64 + (c - 16) * 8;
#pragma unroll
    for (int j = 0; j < 8; ++j) val[j] = (__bf16)ep[j];
  }
  *(v8bf*)(z + (size_t)e * 192 + c * 8) = val;
}

__global__ void k_colstats(const float* __restrict__ X, int M, int N,
                           float* __restrict__ sum, float* __restrict__ sumsq)
{
  const int t   = threadIdx.x;
  const int col = t % N;
  const int rpb = blockDim.x / N;
  float s = 0.f, q = 0.f;
  for (int r = blockIdx.x * rpb + t / N; r < M; r += gridDim.x * rpb) {
    const float v = X[(size_t)r * N + col];
    s += v;
    q += v * v;
  }
  atomicAdd(&sum[col], s);
  atomicAdd(&sumsq[col], q);
}

__global__ void k_bn_finalize(const float* __restrict__ sum,
                              const float* __restrict__ sumsq, int M, int N,
                              float* __restrict__ mu, float* __restrict__ rs)
{
  const int c = blockIdx.x * blockDim.x + threadIdx.x;
  if (c >= N) return;
  const float m = sum[c] / (float)M;
  const float v = sumsq[c] / (float)M - m * m;
  mu[c] = m;
  rs[c] = rsqrtf(v + BN_EPS);
}

// ek = bn(h); edge += ek; z[:,128:192] = bf16(ek); 8 dims per thread
__global__ void k_apply_bn_edge(const float* __restrict__ h,
                                const float* __restrict__ mu,
                                const float* __restrict__ rs,
                                const float* __restrict__ g,
                                const float* __restrict__ b,
                                float* __restrict__ edgef,
                                __bf16* __restrict__ z, int ne)
{
  const int i = blockIdx.x * blockDim.x + threadIdx.x;  // ne * 8
  if (i >= ne * 8) return;
  const int e = i >> 3, c = (i & 7) * 8;
  const float* hp = h + (size_t)e * 64 + c;
  float* ep = edgef + (size_t)e * 64 + c;
  v8bf zb;
#pragma unroll
  for (int j = 0; j < 8; ++j) {
    const int d = c + j;
    const float ek = (hp[j] - mu[d]) * rs[d] * g[d] + b[d];
    ep[j] += ek;
    zb[j] = (__bf16)ek;
  }
  *(v8bf*)(z + (size_t)e * 192 + 128 + c) = zb;
}

__device__ __forceinline__ unsigned f2ord(float f) {
  const unsigned u = __float_as_uint(f);
  return (u & 0x80000000u) ? ~u : (u | 0x80000000u);
}
__device__ __forceinline__ float ord2f(unsigned k) {
  return __uint_as_float((k & 0x80000000u) ? (k & 0x7fffffffu) : ~k);
}

__global__ void k_seg_max(const float* __restrict__ sij,
                          const int* __restrict__ idx1,
                          unsigned* __restrict__ nmax, int ne)
{
  const int i = blockIdx.x * blockDim.x + threadIdx.x;
  if (i >= ne * 128) return;
  const int e = i >> 7, d = i & 127;
  atomicMax(&nmax[(size_t)idx1[e] * 128 + d], f2ord(sij[i]));
}

__global__ void k_seg_expsum(float* __restrict__ sij,
                             const int* __restrict__ idx1,
                             const unsigned* __restrict__ nmax,
                             float* __restrict__ denom, int ne)
{
  const int i = blockIdx.x * blockDim.x + threadIdx.x;
  if (i >= ne * 128) return;
  const int e = i >> 7, d = i & 127;
  const size_t gi = (size_t)idx1[e] * 128 + d;
  const float ev = __expf(sij[i] - ord2f(nmax[gi]));
  sij[i] = ev;
  atomicAdd(&denom[gi], ev);
}

__global__ void k_seg_msg(const float* __restrict__ evals,
                          const __bf16* __restrict__ mij,
                          const int* __restrict__ idx1,
                          const float* __restrict__ denom,
                          float* __restrict__ msg, int ne)
{
  const int i = blockIdx.x * blockDim.x + threadIdx.x;
  if (i >= ne * 128) return;
  const int e = i >> 7, d = i & 127;
  const size_t gi = (size_t)idx1[e] * 128 + d;
  atomicAdd(&msg[gi], (evals[i] / denom[gi]) * (float)mij[i]);
}

__global__ void k_apply_bn_node(const float* __restrict__ msg2,
                                const float* __restrict__ mu,
                                const float* __restrict__ rs,
                                const float* __restrict__ g,
                                const float* __restrict__ b,
                                float* __restrict__ nodef,
                                __bf16* __restrict__ nodeb, int nn)
{
  const int i = blockIdx.x * blockDim.x + threadIdx.x;
  if (i >= nn * 64) return;
  const int d = i & 63;
  const float nv = nodef[i] + (msg2[i] - mu[d]) * rs[d] * g[d] + b[d];
  nodef[i] = nv;
  nodeb[i] = (__bf16)nv;
}

__global__ void k_edge_accum(const float* __restrict__ edgef,
                             const int* __restrict__ idx1,
                             float* __restrict__ esum,
                             float* __restrict__ ecnt, int ne)
{
  const int i = blockIdx.x * blockDim.x + threadIdx.x;
  if (i >= ne * 64) return;
  const int e = i >> 6, d = i & 63;
  atomicAdd(&esum[(size_t)idx1[e] * 64 + d], edgef[i]);
  if (d == 0) atomicAdd(&ecnt[idx1[e]], 1.f);
}

__global__ void k_build_crysN(const float* __restrict__ esum,
                              const float* __restrict__ ecnt,
                              const float* __restrict__ nodef,
                              float* __restrict__ crysN, int nn)
{
  const int i = blockIdx.x * blockDim.x + threadIdx.x;
  if (i >= nn * 128) return;
  const int v = i >> 7, d = i & 127;
  crysN[i] = (d < 64) ? esum[(size_t)v * 64 + d] / fmaxf(ecnt[v], 1.f)
                      : nodef[(size_t)v * 64 + (d - 64)];
}

__global__ void k_graph_accum(const float* __restrict__ crysN,
                              const int* __restrict__ idx3,
                              float* __restrict__ gsum,
                              float* __restrict__ gcnt, int nn)
{
  const int i = blockIdx.x * blockDim.x + threadIdx.x;
  if (i >= nn * 128) return;
  const int v = i >> 7, d = i & 127;
  const int g = idx3[v];
  atomicAdd(&gsum[(size_t)g * 128 + d], crysN[i]);
  if (d == 0) atomicAdd(&gcnt[g], 1.f);
}

__global__ void k_graph_div(const float* __restrict__ gsum,
                            const float* __restrict__ gcnt,
                            __bf16* __restrict__ crys, int ng, int ngp)
{
  const int i = blockIdx.x * blockDim.x + threadIdx.x;
  if (i >= ngp * 128) return;
  const int g = i >> 7;
  crys[i] = (g < ng) ? (__bf16)(gsum[i] / fmaxf(gcnt[g], 1.f)) : (__bf16)0.0f;
}

// ---------------------------------------------------------------------------
// Host orchestration
// ---------------------------------------------------------------------------
static inline int ceil_div(int a, int b) { return (a + b - 1) / b; }

struct LayerP {
  const float *WoutW, *Woutb, *bn1b, *bn1g, *bn2b, *bn2g;
  const float *aW1, *ab1, *aW2, *ab2;
  const float *mW1, *mb1, *mW2, *mb2;
  const float *pW1, *pb1, *pW2, *pb2, *pW3, *pb3;
};

extern "C" void kernel_launch(void* const* d_in, const int* in_sizes, int n_in,
                              void* d_out, int out_size, void* d_ws,
                              size_t ws_size, hipStream_t stream)
{
  (void)n_in; (void)ws_size;
  // ---- input mapping: JAX pytree flattening (dicts in sorted-key order) ----
  const float* edge_fea = (const float*)d_in[0];
  const int*   idx1     = (const int*)d_in[1];
  const int*   idx2     = (const int*)d_in[2];
  const int*   idx3     = (const int*)d_in[3];
  const int*   node_fea = (const int*)d_in[4];
  const float* cW = (const float*)d_in[5];   // [128,256]
  const float* cb = (const float*)d_in[6];
  const float* eW = (const float*)d_in[7];   // [50,64]
  const float* eb = (const float*)d_in[8];
  const float* oW = (const float*)d_in[9];   // [256,2]
  const float* ob = (const float*)d_in[10];
  const float* fW = (const float*)d_in[11];  // [256,256]
  const float* fb = (const float*)d_in[12];
  LayerP L[2];
  for (int l = 0; l < 2; ++l) {
    const int B = 13 + l * 20;
    L[l].WoutW = (const float*)d_in[B + 0];
    L[l].Woutb = (const float*)d_in[B + 1];
    L[l].bn1b  = (const float*)d_in[B + 2];
    L[l].bn1g  = (const float*)d_in[B + 3];
    L[l].bn2b  = (const float*)d_in[B + 4];
    L[l].bn2g  = (const float*)d_in[B + 5];
    L[l].aW1   = (const float*)d_in[B + 6];
    L[l].ab1   = (const float*)d_in[B + 7];
    L[l].aW2   = (const float*)d_in[B + 8];
    L[l].ab2   = (const float*)d_in[B + 9];
    L[l].mW1   = (const float*)d_in[B + 10];
    L[l].mb1   = (const float*)d_in[B + 11];
    L[l].mW2   = (const float*)d_in[B + 12];
    L[l].mb2   = (const float*)d_in[B + 13];
    L[l].pW1   = (const float*)d_in[B + 14];
    L[l].pb1   = (const float*)d_in[B + 15];
    L[l].pW2   = (const float*)d_in[B + 16];
    L[l].pb2   = (const float*)d_in[B + 17];
    L[l].pW3   = (const float*)d_in[B + 18];
    L[l].pb3   = (const float*)d_in[B + 19];
  }
  const float* v_emb = (const float*)d_in[53];  // [100,64]

  const int NE  = in_sizes[1];             // 400000 (multiple of 32)
  const int NN  = in_sizes[3];             // 25000
  const int NNp = (NN + 31) & ~31;         // 25024
  const int NG  = out_size / 2;            // 250
  const int NGp = (NG + 31) & ~31;         // 256

  // ---- workspace bump allocator (256B aligned) -----------------------------
  size_t off = 0;
  auto alloc = [&](size_t bytes) -> void* {
    void* p = (char*)d_ws + off;
    off = (off + bytes + 255) & ~(size_t)255;
    return p;
  };
  __bf16*   efp   = (__bf16*)  alloc((size_t)NE * 64 * 2);
  __bf16*   z     = (__bf16*)  alloc((size_t)NE * 192 * 2);
  __bf16*   t12   = (__bf16*)  alloc((size_t)NE * 128 * 2);  // halves: t1|t2
  float*    h     = (float*)   alloc((size_t)NE * 64 * 4);
  float*    edgef = (float*)   alloc((size_t)NE * 64 * 4);
  float*    sij   = (float*)   alloc((size_t)NE * 128 * 4);
  __bf16*   mij   = (__bf16*)  alloc((size_t)NE * 128 * 2);
  float*    nodef = (float*)   alloc((size_t)NN * 64 * 4);
  __bf16*   nodeb = (__bf16*)  alloc((size_t)NN * 64 * 2);
  unsigned* nmax  = (unsigned*)alloc((size_t)NN * 128 * 4);
  float*    denom = (float*)   alloc((size_t)NN * 128 * 4);
  float*    msg   = (float*)   alloc((size_t)NN * 128 * 4);
  __bf16*   msgb  = (__bf16*)  alloc((size_t)NNp * 128 * 2);
  float*    msg2  = (float*)   alloc((size_t)NN * 64 * 4);
  float*    stat_s= (float*)   alloc(128 * 4);
  float*    stat_q= (float*)   alloc(128 * 4);
  float*    bn_mu = (float*)   alloc(128 * 4);
  float*    bn_rs = (float*)   alloc(128 * 4);
  float*    bam1  = (float*)   alloc(128 * 4);   // concat(ab1, mb1)
  float*    esum  = (float*)   alloc((size_t)NN * 64 * 4);
  float*    ecnt  = (float*)   alloc((size_t)NN * 4);
  float*    crysN = (float*)   alloc((size_t)NN * 128 * 4);
  float*    gsum  = (float*)   alloc((size_t)NG * 128 * 4);
  float*    gcnt  = (float*)   alloc((size_t)NG * 4);
  __bf16*   cry0  = (__bf16*)  alloc((size_t)NGp * 128 * 2);
  __bf16*   cry1  = (__bf16*)  alloc((size_t)NGp * 256 * 2);
  __bf16*   cry2  = (__bf16*)  alloc((size_t)NGp * 256 * 2);

  const int TPB = 256;
#define EW(n) <<<ceil_div((n), TPB), TPB, 0, stream>>>
  auto gemm_grid = [](int M, int N) {
    return dim3(ceil_div(M, 128), ceil_div(N, 64));
  };
  auto pack_w = [&](const float* W, int K, int N) -> __bf16* {
    const int Kb = ceil_div(K, 32), Nb = ceil_div(N, 64) * 4;
    const int el = Kb * Nb * 512;
    __bf16* p = (__bf16*)alloc((size_t)el * 2);
    k_pack_w EW(el)(W, K, N, p, Kb, Nb);
    return p;
  };

  __bf16* eWp = pack_w(eW, 50, 64);
  __bf16* cWp = pack_w(cW, 128, 256);
  __bf16* fWp = pack_w(fW, 256, 256);
  __bf16* oWp = pack_w(oW, 256, 2);
  __bf16 *Pp1[2], *Pp2[2], *Pp3[2], *Pam1[2], *Pa2[2], *Pm2[2], *Pwo[2];
  for (int l = 0; l < 2; ++l) {
    Pp1[l] = pack_w(L[l].pW1, 192, 64);
    Pp2[l] = pack_w(L[l].pW2, 64, 64);
    Pp3[l] = pack_w(L[l].pW3, 64, 64);
    // fused [aW1 | mW1] -> [192,128]: nb-major layout, halves concatenate
    Pam1[l] = (__bf16*)alloc((size_t)8 * 6 * 512 * 2);
    k_pack_w EW(4 * 6 * 512)(L[l].aW1, 192, 64, Pam1[l], 6, 4);
    k_pack_w EW(4 * 6 * 512)(L[l].mW1, 192, 64, Pam1[l] + 4 * 6 * 512, 6, 4);
    Pa2[l] = pack_w(L[l].aW2, 64, 128);
    Pm2[l] = pack_w(L[l].mW2, 64, 128);
    Pwo[l] = pack_w(L[l].WoutW, 128, 64);
  }

  // ---- embeddings ----------------------------------------------------------
  k_embed_nodes EW(NN * 8)(node_fea, v_emb, nodef, nodeb, NN);
  k_cvt_pad EW(NE * 64)(edge_fea, efp, NE, 50, NE, 64);
  gemm_bf16<0, 0, 64, 64><<<gemm_grid(NE, 64), 128, 0, stream>>>(
      efp, eWp, eb, edgef, NE, 64, 64);

  // ---- attention layers ----------------------------------------------------
  for (int l = 0; l < 2; ++l) {
    const LayerP& P = L[l];
    k_build_z EW(NE * 24)(nodeb, edgef, idx1, idx2, z, NE);

    // phi_e: 192->64 (lrelu) -> 64 (lrelu) -> 64   (through t12 halves)
    gemm_bf16<1, 1, 192, 192><<<gemm_grid(NE, 64), 128, 0, stream>>>(
        z, Pp1[l], P.pb1, t12, NE, 64, 128);
    gemm_bf16<1, 1, 64, 128><<<gemm_grid(NE, 64), 128, 0, stream>>>(
        t12, Pp2[l], P.pb2, t12 + 64, NE, 64, 128);
    gemm_bf16<0, 0, 64, 128><<<gemm_grid(NE, 64), 128, 0, stream>>>(
        t12 + 64, Pp3[l], P.pb3, h, NE, 64, 64);

    // bn1, edge += ek, z[:,128:] = ek
    hipMemsetAsync(stat_s, 0, 64 * 4, stream);
    hipMemsetAsync(stat_q, 0, 64 * 4, stream);
    k_colstats<<<512, 256, 0, stream>>>(h, NE, 64, stat_s, stat_q);
    k_bn_finalize<<<1, 64, 0, stream>>>(stat_s, stat_q, NE, 64, bn_mu, bn_rs);
    k_apply_bn_edge EW(NE * 8)(h, bn_mu, bn_rs, P.bn1g, P.bn1b, edgef, z, NE);

    // fused stage-1 of fcnna+fcnnm: one pass over z2 -> t12 = [ta | tm]
    hipMemcpyAsync(bam1, P.ab1, 64 * 4, hipMemcpyDeviceToDevice, stream);
    hipMemcpyAsync(bam1 + 64, P.mb1, 64 * 4, hipMemcpyDeviceToDevice, stream);
    gemm_bf16<1, 1, 192, 192><<<gemm_grid(NE, 128), 128, 0, stream>>>(
        z, Pam1[l], bam1, t12, NE, 128, 128);
    // stage-2: sij (f32, for softmax) and mij (bf16)
    gemm_bf16<0, 0, 64, 128><<<gemm_grid(NE, 128), 128, 0, stream>>>(
        t12, Pa2[l], P.ab2, sij, NE, 128, 128);
    gemm_bf16<1, 0, 64, 128><<<gemm_grid(NE, 128), 128, 0, stream>>>(
        t12 + 64, Pm2[l], P.mb2, mij, NE, 128, 128);

    // segment softmax over idx1 + weighted message sum
    hipMemsetAsync(nmax,  0, (size_t)NN * 128 * 4, stream);  // 0 == key(-inf)
    hipMemsetAsync(denom, 0, (size_t)NN * 128 * 4, stream);
    hipMemsetAsync(msg,   0, (size_t)NN * 128 * 4, stream);
    k_seg_max    EW(NE * 128)(sij, idx1, nmax, NE);
    k_seg_expsum EW(NE * 128)(sij, idx1, nmax, denom, NE);
    k_seg_msg    EW(NE * 128)(sij, mij, idx1, denom, msg, NE);

    // Wout + bn2 + residual node update
    k_cvt_pad EW(NNp * 128)(msg, msgb, NN, 128, NNp, 128);
    gemm_bf16<0, 0, 128, 128><<<gemm_grid(NN, 64), 128, 0, stream>>>(
        msgb, Pwo[l], P.Woutb, msg2, NN, 64, 64);
    hipMemsetAsync(stat_s, 0, 64 * 4, stream);
    hipMemsetAsync(stat_q, 0, 64 * 4, stream);
    k_colstats<<<512, 256, 0, stream>>>(msg2, NN, 64, stat_s, stat_q);
    k_bn_finalize<<<1, 64, 0, stream>>>(stat_s, stat_q, NN, 64, bn_mu, bn_rs);
    k_apply_bn_node EW(NN * 64)(msg2, bn_mu, bn_rs, P.bn2g, P.bn2b, nodef,
                                nodeb, NN);
  }

  // ---- readout -------------------------------------------------------------
  hipMemsetAsync(esum, 0, (size_t)NN * 64 * 4, stream);
  hipMemsetAsync(ecnt, 0, (size_t)NN * 4, stream);
  k_edge_accum  EW(NE * 64)(edgef, idx1, esum, ecnt, NE);
  k_build_crysN EW(NN * 128)(esum, ecnt, nodef, crysN, NN);

  hipMemsetAsync(gsum, 0, (size_t)NG * 128 * 4, stream);
  hipMemsetAsync(gcnt, 0, (size_t)NG * 4, stream);
  k_graph_accum EW(NN * 128)(crysN, idx3, gsum, gcnt, NN);
  k_graph_div   EW(NGp * 128)(gsum, gcnt, cry0, NG, NGp);

  gemm_bf16<1, 1, 128, 128><<<gemm_grid(NG, 256), 128, 0, stream>>>(
      cry0, cWp, cb, cry1, NG, 256, 256);
  gemm_bf16<1, 1, 256, 256><<<gemm_grid(NG, 256), 128, 0, stream>>>(
      cry1, fWp, fb, cry2, NG, 256, 256);
  gemm_bf16<0, 0, 256, 256><<<gemm_grid(NG, 64), 128, 0, stream>>>(
      cry2, oWp, ob, (float*)d_out, NG, 2, 2);
#undef EW
}
